// Feature_extraction_layer_15049565405702
// MI455X (gfx1250) — matrix operand order repaired and checked by reading, not compiled, not run on hardware
//
#include <hip/hip_runtime.h>
#include <hip/hip_bf16.h>
#include <math.h>

typedef __attribute__((ext_vector_type(16))) _Float16 v16h;
typedef __attribute__((ext_vector_type(8)))  float    v8f;

#define BATCH 2
#define CH    36
#define NPTS  8192
#define INCH  32
#define OUTCH 64
#define KNN   4
#define H1    256
#define HA    128
#define OCH   68      // 4 passthrough + 64

#define PTS_PER_BLK 4
#define NBR 5
#define VROWS (PTS_PER_BLK * NBR)   // 20 valid rows
#define RP 32                        // padded to 2 M-tiles

// KNN blocking
#define QPB   64                     // queries per block (4 N-tiles)
#define CCH   128                    // candidates staged per chunk (8 M-tiles)
#define NCHUNK (NPTS / CCH)          // 64

// ---- fragment loaders (CDNA5 wave32 WMMA 16x16x32 f16 layouts) ----
// A (16x32 f16): lane l -> row M=l%16; hi=l/16 -> K in {8*hi..8*hi+7} U {16+8*hi..+7}
static __device__ __forceinline__ v16h load_Afrag(const _Float16* row, int hi) {
  v16h a;
  *(float4*)&a         = *(const float4*)(row + 8 * hi);
  *(((float4*)&a) + 1) = *(const float4*)(row + 16 + 8 * hi);
  return a;
}
// B (32x16 f16), stored as B^T rows (row n = column n of B): lane l -> N=l%16,
// K = kb + {8*hi..8*hi+7} U {kb+16+8*hi..+7}
static __device__ __forceinline__ v16h load_Bfrag(const _Float16* btrow, int kb, int hi) {
  v16h b;
  *(float4*)&b         = *(const float4*)(btrow + kb + 8 * hi);
  *(((float4*)&b) + 1) = *(const float4*)(btrow + kb + 16 + 8 * hi);
  return b;
}
// A fragment sourced from an f32 row (convert inline)
static __device__ __forceinline__ v16h cvt_Afrag(const float* row, int kb, int hi) {
  v16h a;
  #pragma unroll
  for (int i = 0; i < 8; ++i) a[i] = (_Float16)row[kb + 8 * hi + i];
  #pragma unroll
  for (int i = 0; i < 8; ++i) a[8 + i] = (_Float16)row[kb + 16 + 8 * hi + i];
  return a;
}
static __device__ __forceinline__ float gelu_exact(float x) {
  return 0.5f * x * (1.0f + erff(x * 0.70710678118654752f));
}
// async global->LDS copy (ASYNCcnt path); 16 bytes per lane
static __device__ __forceinline__ void async_copy_b128(void* lds_dst,
                                                       const void* gsrc) {
  unsigned l = (unsigned)(uintptr_t)lds_dst;   // flat[31:0] == LDS offset
  unsigned long long g = (unsigned long long)(uintptr_t)gsrc;
  asm volatile("global_load_async_to_lds_b128 %0, %1, off"
               :: "v"(l), "v"(g) : "memory");
}
static __device__ __forceinline__ void wait_async0() {
  asm volatile("s_wait_asynccnt 0x0" ::: "memory");
}

// ---------------- prep: x (B,36,P) -> f16 rows (B,P,32) + |f|^2 ----------------
__global__ void __launch_bounds__(256) k_prep(const float* __restrict__ x,
                                              _Float16* __restrict__ fh,
                                              float* __restrict__ sq) {
  int gid = blockIdx.x * 256 + threadIdx.x;  // over B*NPTS
  int b = gid / NPTS, p = gid % NPTS;
  const float* xb = x + (size_t)b * CH * NPTS + p;
  _Float16* row = fh + (size_t)(b * NPTS + p) * INCH;
  float s = 0.f;
  #pragma unroll
  for (int d = 0; d < INCH; ++d) {
    float v = xb[(size_t)(4 + d) * NPTS];
    s += v * v;
    row[d] = (_Float16)v;
  }
  sq[b * NPTS + p] = s;
}

// ---------------- weight transpose/convert to f16 B^T layouts ----------------
__global__ void __launch_bounds__(256) k_wcvt(const float* __restrict__ W1,
                                              const float* __restrict__ W2,
                                              const float* __restrict__ Wa1,
                                              _Float16* __restrict__ W1t,
                                              _Float16* __restrict__ W2t,
                                              _Float16* __restrict__ Wa1t) {
  int gid = blockIdx.x * 256 + threadIdx.x;
  if (gid < H1 * INCH) {                       // W1t[n][k] = W1[k][n]
    int n = gid / INCH, k = gid % INCH;
    W1t[gid] = (_Float16)W1[k * H1 + n];
  } else if (gid < H1 * INCH + OUTCH * H1) {   // W2t[n][k] = W2[k][n]
    int e = gid - H1 * INCH;
    int n = e / H1, k = e % H1;
    W2t[e] = (_Float16)W2[k * OUTCH + n];
  } else if (gid < H1 * INCH + OUTCH * H1 + HA * OUTCH) {  // Wa1t[n][k] = Wa1[k][n]
    int e = gid - H1 * INCH - OUTCH * H1;
    int n = e / OUTCH, k = e % OUTCH;
    Wa1t[e] = (_Float16)Wa1[k * HA + n];
  }
}

// ---------------- KNN: async-LDS candidate chunks + WMMA + per-lane top-4 ----
// Block: 8 waves, 64 queries (4 query tiles; waves {qt, qt+4} share tile qt and
// split the 8 candidate M-tiles of each chunk). Candidate rows are staged into
// a double-buffered LDS tile with GLOBAL_LOAD_ASYNC_TO_LDS_B128 (ASYNCcnt),
// so each block reads fh exactly once and staging overlaps the WMMA+select
// phase without burning VGPRs. Selection key = |c|^2 - 2*c.q (the per-query
// constant |q|^2 is ordering-invariant; self remains minimal by Cauchy-Schwarz).
__global__ void __launch_bounds__(256) k_knn(const _Float16* __restrict__ fh,
                                             const float* __restrict__ sq,
                                             int* __restrict__ knn_idx) {
  __shared__ _Float16 s_cand[2][CCH * INCH]; // 2 x 8 KB candidate chunks
  __shared__ float md[QPB * 16];             // 4 KB merge keys
  __shared__ int   mi[QPB * 16];             // 4 KB merge indices

  const int wave = threadIdx.x >> 5;
  const int lane = threadIdx.x & 31;
  const int lo16 = lane & 15;
  const int hi   = lane >> 4;
  const int qt   = wave & 3;        // query tile 0..3
  const int mg   = wave >> 2;       // candidate M-tile group 0..1 (4 tiles each)

  const int blocks_per_b = NPTS / QPB;
  const int b  = blockIdx.x / blocks_per_b;
  const int qb = (blockIdx.x % blocks_per_b) * QPB;
  const _Float16* fb = fh + (size_t)b * NPTS * INCH;
  const float*    sb = sq + (size_t)b * NPTS;

  // query B-fragment (column N = query), resident for the whole sweep
  const int q = qb + qt * 16 + lo16;
  const v16h bq = load_Bfrag(fb + (size_t)q * INCH, 0, hi);

  float bd0 = 3.4e38f, bd1 = 3.4e38f, bd2 = 3.4e38f, bd3 = 3.4e38f;
  int   bi0 = 0, bi1 = 0, bi2 = 0, bi3 = 0;

  // per-thread staging slice: 2 x 16B of the 8 KB chunk
  const int srow = threadIdx.x >> 2;        // 0..63
  const int spart = (threadIdx.x & 3) * 8;  // 0,8,16,24 halves

  // prologue: async-fill buffer 0 with chunk 0
  {
    const _Float16* g0 = fb + (size_t)srow * INCH + spart;
    async_copy_b128(&s_cand[0][srow * INCH + spart], g0);
    async_copy_b128(&s_cand[0][(srow + 64) * INCH + spart], g0 + 64 * INCH);
  }
  wait_async0();
  __syncthreads();

  for (int c = 0; c < NCHUNK; ++c) {
    const int cur = c & 1;
    if (c + 1 < NCHUNK) {                   // async-fill the other buffer
      const _Float16* g0 = fb + (size_t)(c + 1) * CCH * INCH +
                           (size_t)srow * INCH + spart;
      async_copy_b128(&s_cand[cur ^ 1][srow * INCH + spart], g0);
      async_copy_b128(&s_cand[cur ^ 1][(srow + 64) * INCH + spart],
                      g0 + 64 * INCH);
    }
    if (c + 2 < NCHUNK) {                   // speculative prefetch 2 ahead
      __builtin_prefetch(fb + (size_t)(c + 2) * CCH * INCH +
                             (size_t)srow * INCH, 0, 1);
    }

    const int cb = c * CCH;
    #pragma unroll
    for (int mt = 0; mt < 4; ++mt) {
      const int mrow = mg * 64 + mt * 16;   // M-tile base within chunk
      v16h av = load_Afrag(s_cand[cur] + (mrow + lo16) * INCH, hi);
      v8f acc = {};
      acc = __builtin_amdgcn_wmma_f32_16x16x32_f16(false, av, false, bq,
                                                   (short)0, acc, false, false);
      // C layout: lane -> N=lo16 (query), M = 8*hi + r (candidate in tile)
      const int cbase = cb + mrow + 8 * hi;
      float4 n0 = *(const float4*)(sb + cbase);
      float4 n1 = *(const float4*)(sb + cbase + 4);
      float sc[8] = { n0.x, n0.y, n0.z, n0.w, n1.x, n1.y, n1.z, n1.w };
      #pragma unroll
      for (int r = 0; r < 8; ++r) {
        float v = sc[r] - 2.0f * acc[r];    // |q|^2 dropped (order-invariant)
        int ci = cbase + r;
        if (v < bd3) {
          if (v < bd2) {
            bd3 = bd2; bi3 = bi2;
            if (v < bd1) {
              bd2 = bd1; bi2 = bi1;
              if (v < bd0) { bd1 = bd0; bi1 = bi0; bd0 = v; bi0 = ci; }
              else         { bd1 = v;   bi1 = ci; }
            } else { bd2 = v; bi2 = ci; }
          } else { bd3 = v; bi3 = ci; }
        }
      }
    }

    if (c + 1 < NCHUNK) wait_async0();      // next buffer complete
    __syncthreads();                        // all waves done with cur + next ready
  }

  // merge: per query, 2 waves x 2 half-waves x top4 = 16 partials
  const int qlocal = qt * 16 + lo16;                // 0..63
  const int eg = mg * 2 + hi;                       // entry group 0..3
  int slot = qlocal * 16 + eg * 4;
  md[slot + 0] = bd0; md[slot + 1] = bd1; md[slot + 2] = bd2; md[slot + 3] = bd3;
  mi[slot + 0] = bi0; mi[slot + 1] = bi1; mi[slot + 2] = bi2; mi[slot + 3] = bi3;
  __syncthreads();
  if (threadIdx.x < QPB) {
    float* rd = md + threadIdx.x * 16;
    int*   ri = mi + threadIdx.x * 16;
    int* outp = knn_idx + (size_t)(b * NPTS + qb + threadIdx.x) * KNN;
    for (int j = 0; j < KNN; ++j) {
      float best = rd[0]; int bidx = ri[0], bslot = 0;
      #pragma unroll
      for (int e = 1; e < 16; ++e)
        if (rd[e] < best) { best = rd[e]; bidx = ri[e]; bslot = e; }
      rd[bslot] = 3.4e38f;
      outp[j] = bidx;
    }
  }
}

// ---------------- fused neighbor-MLP + attention pool + residual ----------------
__global__ void __launch_bounds__(256) k_mlp(
    const float* __restrict__ x,
    const _Float16* __restrict__ fh,
    const int* __restrict__ knn_idx,
    const _Float16* __restrict__ W1t, const float* __restrict__ b1,
    const float* __restrict__ g1, const float* __restrict__ be1,
    const _Float16* __restrict__ W2t, const float* __restrict__ b2,
    const _Float16* __restrict__ Wa1t, const float* __restrict__ ba1,
    const float* __restrict__ ga, const float* __restrict__ bba,
    const float* __restrict__ Wa2, const float* __restrict__ ba2,
    const float* __restrict__ We, const float* __restrict__ bee,
    float* __restrict__ out) {
  __shared__ _Float16 s_feats[RP * INCH];   // 2 KB
  __shared__ float s_t1[RP * H1];           // 32 KB
  __shared__ float s_h[RP * OUTCH];         // 8 KB
  __shared__ float s_att[RP * HA];          // 16 KB
  __shared__ float s_logit[RP];
  __shared__ float s_aw[RP];

  const int wave = threadIdx.x >> 5;
  const int lane = threadIdx.x & 31;
  const int lo16 = lane & 15;
  const int hi   = lane >> 4;
  const int blocks_per_b = NPTS / PTS_PER_BLK;
  const int b  = blockIdx.x / blocks_per_b;
  const int pb = (blockIdx.x % blocks_per_b) * PTS_PER_BLK;

  // gather 5 feature rows per point (4 KNN + self), zero-pad rows >= 20
  for (int e = threadIdx.x; e < RP * INCH; e += 256) {
    int row = e / INCH, k = e % INCH;
    _Float16 v = (_Float16)0.0f;
    if (row < VROWS) {
      int p = row / NBR, n = row % NBR;
      int pg = pb + p;
      int src = (n == NBR - 1) ? pg
                               : knn_idx[(size_t)(b * NPTS + pg) * KNN + n];
      v = fh[(size_t)(b * NPTS + src) * INCH + k];
    }
    s_feats[e] = v;
  }
  __syncthreads();

  // GEMM1: (32x32) @ (32x256), 2 x 16 tiles, 4 per wave
  for (int t = wave; t < 2 * (H1 / 16); t += 8) {
    int mt = t & 1, nt = t >> 1;
    v16h a = load_Afrag(s_feats + (mt * 16 + lo16) * INCH, hi);
    v16h w = load_Bfrag(W1t + (size_t)(nt * 16 + lo16) * INCH, 0, hi);
    v8f acc = {};
    acc = __builtin_amdgcn_wmma_f32_16x16x32_f16(false, a, false, w,
                                                 (short)0, acc, false, false);
    int col = nt * 16 + lo16;
    float bias = b1[col];
    #pragma unroll
    for (int r = 0; r < 8; ++r)
      s_t1[(mt * 16 + 8 * hi + r) * H1 + col] = acc[r] + bias;
  }
  __syncthreads();

  // LayerNorm(256) + exact GELU, in place (one thread per row, race-free)
  if (threadIdx.x < VROWS) {
    float* row = s_t1 + threadIdx.x * H1;
    float m = 0.f;
    for (int i = 0; i < H1; ++i) m += row[i];
    m *= (1.0f / H1);
    float var = 0.f;
    for (int i = 0; i < H1; ++i) { float d = row[i] - m; var += d * d; }
    var *= (1.0f / H1);
    float inv = rsqrtf(var + 1e-5f);
    for (int i = 0; i < H1; ++i)
      row[i] = gelu_exact((row[i] - m) * inv * g1[i] + be1[i]);
  }
  __syncthreads();

  // GEMM2: (32x256) @ (256x64), 2 x 4 tiles, 1 per wave, K-accumulate 8 steps
  {
    int mt = wave & 1, nt = wave >> 1;
    v8f acc = {};
    #pragma unroll
    for (int ks = 0; ks < H1 / 32; ++ks) {
      v16h a = cvt_Afrag(s_t1 + (mt * 16 + lo16) * H1, ks * 32, hi);
      v16h w = load_Bfrag(W2t + (size_t)(nt * 16 + lo16) * H1, ks * 32, hi);
      acc = __builtin_amdgcn_wmma_f32_16x16x32_f16(false, a, false, w,
                                                   (short)0, acc, false, false);
    }
    int col = nt * 16 + lo16;
    float bias = b2[col];
    #pragma unroll
    for (int r = 0; r < 8; ++r)
      s_h[(mt * 16 + 8 * hi + r) * OUTCH + col] = acc[r] + bias;
  }
  __syncthreads();

  // GEMM3 (attention): (32x64) @ (64x128), 2 x 8 tiles, 2 per wave, K = 2 steps
  for (int t = wave; t < 2 * (HA / 16); t += 8) {
    int mt = t & 1, nt = t >> 1;
    v8f acc = {};
    #pragma unroll
    for (int ks = 0; ks < OUTCH / 32; ++ks) {
      v16h a = cvt_Afrag(s_h + (mt * 16 + lo16) * OUTCH, ks * 32, hi);
      v16h w = load_Bfrag(Wa1t + (size_t)(nt * 16 + lo16) * OUTCH, ks * 32, hi);
      acc = __builtin_amdgcn_wmma_f32_16x16x32_f16(false, a, false, w,
                                                   (short)0, acc, false, false);
    }
    int col = nt * 16 + lo16;
    float bias = ba1[col];
    #pragma unroll
    for (int r = 0; r < 8; ++r)
      s_att[(mt * 16 + 8 * hi + r) * HA + col] = acc[r] + bias;
  }
  __syncthreads();

  // per-row: LN(128) + GELU + dot(Wa2) -> attention logit
  if (threadIdx.x < VROWS) {
    float* row = s_att + threadIdx.x * HA;
    float m = 0.f;
    for (int i = 0; i < HA; ++i) m += row[i];
    m *= (1.0f / HA);
    float var = 0.f;
    for (int i = 0; i < HA; ++i) { float d = row[i] - m; var += d * d; }
    var *= (1.0f / HA);
    float inv = rsqrtf(var + 1e-5f);
    float acc = 0.f;
    for (int i = 0; i < HA; ++i)
      acc += gelu_exact((row[i] - m) * inv * ga[i] + bba[i]) * Wa2[i];
    s_logit[threadIdx.x] = acc + ba2[0];
  }
  __syncthreads();

  // softmax over the 5 neighbor slots, one thread per point
  if (threadIdx.x < PTS_PER_BLK) {
    const float* lg = s_logit + threadIdx.x * NBR;
    float mx = lg[0];
    for (int n = 1; n < NBR; ++n) mx = fmaxf(mx, lg[n]);
    float s = 0.f, e[NBR];
    for (int n = 0; n < NBR; ++n) { e[n] = __expf(lg[n] - mx); s += e[n]; }
    float invs = 1.0f / s;
    for (int n = 0; n < NBR; ++n) s_aw[threadIdx.x * NBR + n] = e[n] * invs;
  }
  __syncthreads();

  // pooled + residual (f @ We + bee) + transposed store; copy passthrough ch 0..3
  {
    int p = threadIdx.x / OUTCH;   // 0..3
    int d = threadIdx.x % OUTCH;   // 0..63
    int pg = pb + p;
    float pooled = 0.f;
    #pragma unroll
    for (int n = 0; n < NBR; ++n)
      pooled += s_aw[p * NBR + n] * s_h[(p * NBR + n) * OUTCH + d];
    float res = 0.f;
    const _Float16* self = s_feats + (p * NBR + NBR - 1) * INCH;
    #pragma unroll
    for (int k = 0; k < INCH; ++k)
      res += (float)self[k] * We[k * OUTCH + d];
    out[((size_t)b * OCH + 4 + d) * NPTS + pg] = pooled + res + bee[d];
    if (d < 4)
      out[((size_t)b * OCH + d) * NPTS + pg] = x[((size_t)b * CH + d) * NPTS + pg];
  }
}

extern "C" void kernel_launch(void* const* d_in, const int* in_sizes, int n_in,
                              void* d_out, int out_size, void* d_ws, size_t ws_size,
                              hipStream_t stream) {
  (void)in_sizes; (void)n_in; (void)out_size; (void)ws_size;
  const float* x   = (const float*)d_in[0];
  const float* W1  = (const float*)d_in[1];
  const float* b1  = (const float*)d_in[2];
  const float* g1  = (const float*)d_in[3];
  const float* be1 = (const float*)d_in[4];
  const float* W2  = (const float*)d_in[5];
  const float* b2  = (const float*)d_in[6];
  const float* Wa1 = (const float*)d_in[7];
  const float* ba1 = (const float*)d_in[8];
  const float* ga  = (const float*)d_in[9];
  const float* bba = (const float*)d_in[10];
  const float* Wa2 = (const float*)d_in[11];
  const float* ba2 = (const float*)d_in[12];
  const float* We  = (const float*)d_in[13];
  const float* bee = (const float*)d_in[14];
  float* out = (float*)d_out;

  char* ws = (char*)d_ws;
  size_t off = 0;
  _Float16* fh   = (_Float16*)(ws + off); off += (size_t)BATCH * NPTS * INCH * 2;  // 1 MB
  float*    sq   = (float*)(ws + off);    off += (size_t)BATCH * NPTS * 4;          // 64 KB
  _Float16* W1t  = (_Float16*)(ws + off); off += (size_t)H1 * INCH * 2;
  _Float16* W2t  = (_Float16*)(ws + off); off += (size_t)OUTCH * H1 * 2;
  _Float16* Wa1t = (_Float16*)(ws + off); off += (size_t)HA * OUTCH * 2;
  int*      idx  = (int*)(ws + off);      off += (size_t)BATCH * NPTS * KNN * 4;    // 256 KB

  k_prep<<<BATCH * NPTS / 256, 256, 0, stream>>>(x, fh, sq);
  k_wcvt<<<(H1 * INCH + OUTCH * H1 + HA * OUTCH + 255) / 256, 256, 0, stream>>>(
      W1, W2, Wa1, W1t, W2t, Wa1t);
  k_knn<<<BATCH * (NPTS / QPB), 256, 0, stream>>>(fh, sq, idx);
  k_mlp<<<BATCH * (NPTS / PTS_PER_BLK), 256, 0, stream>>>(
      x, fh, idx, W1t, b1, g1, be1, W2t, b2, Wa1t, ba1, ga, bba, Wa2, ba2,
      We, bee, out);
}
